// MolGNN_5231270166914
// MI455X (gfx1250) — matrix operand order repaired and checked by reading, not compiled, not run on hardware
//
#include <hip/hip_runtime.h>
#include <math.h>

typedef __attribute__((ext_vector_type(16))) _Float16 v16h;
typedef __attribute__((ext_vector_type(8)))  float    v8f;
typedef __attribute__((ext_vector_type(4)))  float    v4f;
typedef __attribute__((ext_vector_type(4)))  unsigned int u32x4;
typedef __attribute__((ext_vector_type(4)))  int      i32x4;
typedef __attribute__((ext_vector_type(8)))  int      i32x8;
typedef __attribute__((ext_vector_type(4)))  unsigned int uint4v;

#define P_N  50000
#define M_N  100000
#define T_N  20000
#define D_N  256
#define H_N  256
#define FM_N 512
#define E_N  250000
#define B_N  8192
#define R_N  4

#define XSTR(x) STR(x)
#define STR(x) #x

#if defined(__gfx1250__) && __has_builtin(__builtin_amdgcn_tensor_load_to_lds)
#define HAVE_TDM 1
#else
#define HAVE_TDM 0
#endif

#if defined(__gfx1250__)
#if HAVE_TDM
#pragma message("CDNA5-PROBE: TDM tensor_load_to_lds ENABLED, clang=" XSTR(__clang_major__))
#else
#pragma message("CDNA5-PROBE: TDM builtin NOT available, fallback copy, clang=" XSTR(__clang_major__))
#endif
#endif

#if HAVE_TDM
// One 2D TDM load: (tile_w_u32 x tile_h) 4-byte units, global row stride
// row_stride_u32, into LDS at lds_off. D# packing per ISA 8.3/8.4.
__device__ __forceinline__ void tdm_load_2d(unsigned lds_off, const void* gsrc,
                                            unsigned tile_w_u32, unsigned tile_h,
                                            unsigned row_stride_u32)
{
    const unsigned long long ga = (unsigned long long)(uintptr_t)gsrc;
    u32x4 g0;
    g0[0] = 1u;                                    // count=1, user-mode
    g0[1] = (unsigned)__builtin_amdgcn_readfirstlane((int)lds_off);
    g0[2] = (unsigned)__builtin_amdgcn_readfirstlane((int)(ga & 0xFFFFFFFFu));
    g0[3] = (unsigned)__builtin_amdgcn_readfirstlane(
                (int)(((ga >> 32) & 0x01FFFFFFu) | (2u << 30)));  // addr[56:32] | type=2
    const unsigned td0 = 0x7FFFFFFFu;              // tensor_dim0: huge (no OOB clip)
    i32x8 g1;
    g1[0] = (int)(2u << 16);                       // data_size = 4B
    g1[1] = (int)((td0 & 0xFFFFu) << 16);          // tensor_dim0[15:0]
    g1[2] = (int)(((td0 >> 16) & 0xFFFFu) | ((tile_h & 0xFFFFu) << 16));
    g1[3] = (int)((tile_h >> 16) | ((tile_w_u32 & 0xFFFFu) << 16));
    g1[4] = (int)(tile_h & 0xFFFFu);               // tile_dim1 (tile_dim2=0)
    g1[5] = (int)row_stride_u32;                   // tensor_dim0_stride[31:0]
    g1[6] = 0;
    g1[7] = 0;
    const i32x4 z4 = {};
#if __clang_major__ >= 23
    const i32x8 z8 = {};
    __builtin_amdgcn_tensor_load_to_lds(g0, g1, z4, z4, z8, 0);
#else
    __builtin_amdgcn_tensor_load_to_lds(g0, g1, z4, z4, 0);
#endif
}
#endif

// ---------------------------------------------------------------------------
// Weight pre-pack: W[K,N] f32 -> f16 B fragments in this kernel's k order (matches A frags).
// Layout [kc][nt][lane][h]: lane n = lane%16, k = kc*32 + 16*(lane/16) + h.
// ---------------------------------------------------------------------------
__global__ __launch_bounds__(256)
void k_pack_w(const float* __restrict__ W, _Float16* __restrict__ Wp, int K, int N)
{
    const int idx = blockIdx.x * blockDim.x + threadIdx.x;
    if (idx >= K * N) return;
    const int ntiles = N >> 4;
    const int kc  = idx / (ntiles * 512);
    const int rem = idx - kc * (ntiles * 512);
    const int nt  = rem >> 9;
    const int e   = rem & 511;
    const int ln  = e >> 4;
    const int h   = e & 15;
    const int n = nt * 16 + (ln & 15);
    const int k = kc * 32 + ((ln >> 4) << 4) + h;
    Wp[idx] = (_Float16)W[(size_t)k * N + n];
}

__device__ __forceinline__ v16h load_a_frag(const float* __restrict__ Arow, int kb)
{
    const v4f a0 = *(const v4f*)(Arow + kb);
    const v4f a1 = *(const v4f*)(Arow + kb + 4);
    const v4f a2 = *(const v4f*)(Arow + kb + 8);
    const v4f a3 = *(const v4f*)(Arow + kb + 12);
    v16h a;
#pragma unroll
    for (int q = 0; q < 4; ++q) {
        a[q]      = (_Float16)a0[q];
        a[4 + q]  = (_Float16)a1[q];
        a[8 + q]  = (_Float16)a2[q];
        a[12 + q] = (_Float16)a3[q];
    }
    return a;
}

// ---------------------------------------------------------------------------
// WMMA GEMM:  C[Mr,N] = act( A0@W0 (+ A1@W1) + bias (+ Res) )
// Block = 256 thr = 8 waves; block tile 256(M) x 64(N); each wave owns two
// 16x64 row groups (8 accumulators). Weight panel staged once into dynamic
// LDS via TDM (or coalesced b128 copy); barrier-free k-loop: per 32-K chunk
// 8x global_load_b128 (two A frags) + 4 B frags from LDS -> 8 WMMAs.
// NOTE: row pointers are kept as individual scalars (no pointer arrays) so
// the compiler keeps them in the GLOBAL address space (global_load_b128,
// LOADcnt only) instead of falling back to FLAT (LOADcnt+DScnt).
// ---------------------------------------------------------------------------
__global__ __launch_bounds__(256)
void k_wmma_gemm(const float* __restrict__ A0, const int* __restrict__ idx0, int idx0s, int ld0,
                 const float* __restrict__ A1, const int* __restrict__ idx1, int idx1s, int ld1,
                 const _Float16* __restrict__ Wp0, const _Float16* __restrict__ Wp1,
                 const float* __restrict__ bias, const float* __restrict__ Res,
                 float* __restrict__ C,
                 int Mrows, int N, int K0, int K1, int act)
{
    extern __shared__ v16h panel_v[];   // (Ktot/32) * 128 v16h  (<= 64 KB)

    const int lane = threadIdx.x & 31;
    const int wave = threadIdx.x >> 5;
    const int hl   = lane >> 4;
    const int mloc = lane & 15;
    const int rowBase = blockIdx.y * 256 + wave * 16;   // row group u adds u*128
    const int nt0  = blockIdx.x * 4;
    const int ntiles = N >> 4;
    const int nKc0 = K0 >> 5;
    const int nKc  = (K0 + K1) >> 5;

    // ---- stage weight panel into LDS ----
#if HAVE_TDM
    if (threadIdx.x < 32) {
        const unsigned ldsb = (unsigned)(uintptr_t)panel_v;
        tdm_load_2d(ldsb, Wp0 + (size_t)nt0 * 512,
                    1024, (unsigned)nKc0, (unsigned)(ntiles * 256));
        if (K1 > 0)
            tdm_load_2d(ldsb + (unsigned)nKc0 * 4096u, Wp1 + (size_t)nt0 * 512,
                        1024, (unsigned)(nKc - nKc0), (unsigned)(ntiles * 256));
        __builtin_amdgcn_s_wait_tensorcnt(0);
    }
    __syncthreads();
#else
    for (int c = 0; c < nKc; ++c) {
        const _Float16* src = (c < nKc0)
            ? Wp0 + ((size_t)c * ntiles + nt0) * 512
            : Wp1 + ((size_t)(c - nKc0) * ntiles + nt0) * 512;
        ((uint4v*)(panel_v + c * 128))[threadIdx.x] =
            ((const uint4v*)src)[threadIdx.x];      // 256 thr x 16B = 4KB chunk
    }
    __syncthreads();
#endif

    // ---- resolve (optionally gathered) A row pointers: explicit scalars ----
    int growA = rowBase + mloc;
    if (growA >= Mrows) growA = Mrows - 1;
    int growB = rowBase + 128 + mloc;
    if (growB >= Mrows) growB = Mrows - 1;

    const int r0a = idx0 ? idx0[(size_t)growA * idx0s] : growA;
    const int r0b = idx0 ? idx0[(size_t)growB * idx0s] : growB;
    const float* __restrict__ row0a = A0 + (size_t)r0a * ld0;
    const float* __restrict__ row0b = A0 + (size_t)r0b * ld0;
    const float* __restrict__ row1a = A0;    // harmless defaults (unused if K1==0)
    const float* __restrict__ row1b = A0;
    if (A1) {
        const int r1a = idx1 ? idx1[(size_t)growA * idx1s] : growA;
        const int r1b = idx1 ? idx1[(size_t)growB * idx1s] : growB;
        row1a = A1 + (size_t)r1a * ld1;
        row1b = A1 + (size_t)r1b * ld1;
    }

    v8f acc[2][4];
#pragma unroll
    for (int u = 0; u < 2; ++u)
#pragma unroll
        for (int j = 0; j < 4; ++j) acc[u][j] = (v8f){};

    int c = 0;
    for (int s2 = 0; s2 < 2; ++s2) {
        const float* __restrict__ rA = s2 ? row1a : row0a;
        const float* __restrict__ rB = s2 ? row1b : row0b;
        const int nc = s2 ? (nKc - nKc0) : nKc0;
        for (int i = 0; i < nc; ++i, ++c) {
            const int kb = i * 32 + (hl ? 16 : 0);
            const v16h aA = load_a_frag(rA, kb);
            const v16h aB = load_a_frag(rB, kb);
            if (i + 1 < nc) {
                __builtin_prefetch(rA + kb + 32, 0, 1);
                __builtin_prefetch(rB + kb + 32, 0, 1);
            }
            const v16h* pb = panel_v + c * 128 + lane;
#pragma unroll
            for (int j = 0; j < 4; ++j) {
                const v16h bj = pb[j * 32];
                acc[0][j] = __builtin_amdgcn_wmma_f32_16x16x32_f16(false, aA, false, bj, (short)0, acc[0][j], false, false);
                acc[1][j] = __builtin_amdgcn_wmma_f32_16x16x32_f16(false, aB, false, bj, (short)0, acc[1][j], false, false);
            }
        }
    }

    // ---- epilogue (C/D layout: VGPR g -> row g / g+8; lane%16 -> col) ----
    const int colb = blockIdx.x * 64 + mloc;
#pragma unroll
    for (int u = 0; u < 2; ++u) {
#pragma unroll
        for (int g = 0; g < 8; ++g) {
            const int r = rowBase + u * 128 + g + hl * 8;
            if (r < Mrows) {
                const size_t rb = (size_t)r * N;
#pragma unroll
                for (int j = 0; j < 4; ++j) {
                    const int col = colb + j * 16;
                    float x = acc[u][j][g];
                    if (bias) x += bias[col];
                    if (Res)  x += Res[rb + col];
                    if (act == 1)      x = x > 0.f ? x : 0.f;
                    else if (act == 2) x = x > 0.f ? x : 0.01f * x;
                    C[rb + col] = x;
                }
            }
        }
    }
}

// ---------------------------------------------------------------------------
// Helpers
// ---------------------------------------------------------------------------
__global__ __launch_bounds__(256)
void k_fill(float* __restrict__ p, float v, size_t n)
{
    const size_t i = (size_t)blockIdx.x * blockDim.x + threadIdx.x;
    if (i < n) p[i] = v;
}

// block-per-edge segment_sum: uniform scalar index loads, coalesced row add
__global__ __launch_bounds__(256)
void k_scatter_add(const float* __restrict__ Src, const int* __restrict__ sidx,
                   const int* __restrict__ didx, float* __restrict__ Agg)
{
    const int e = blockIdx.x;
    const int s = sidx[e];      // uniform -> scalar load
    const int d = didx[e];      // uniform -> scalar load
    const int t = threadIdx.x;
    atomicAdd(&Agg[(size_t)d * D_N + t], Src[(size_t)s * D_N + t]);
}

__global__ __launch_bounds__(256)
void k_tanh_comb(const float* __restrict__ a, const float* __restrict__ b,
                 const float* __restrict__ base, float* __restrict__ out,
                 float scale, size_t n)
{
    const size_t i = (size_t)blockIdx.x * blockDim.x + threadIdx.x;
    if (i >= n) return;
    float v = a[i];
    if (b) v += b[i];
    v *= scale;
    out[i] = 0.01f * tanhf(v) + base[i];
}

__global__ __launch_bounds__(256)
void k_layernorm(const float* __restrict__ X, const float* __restrict__ g,
                 const float* __restrict__ b, float* __restrict__ Y)
{
    __shared__ float sm[256];
    const int t = threadIdx.x;
    const size_t base = (size_t)blockIdx.x * 256;
    const float x = X[base + t];
    sm[t] = x; __syncthreads();
    for (int o = 128; o > 0; o >>= 1) { if (t < o) sm[t] += sm[t + o]; __syncthreads(); }
    const float mu = sm[0] * (1.f / 256.f);
    __syncthreads();
    const float d = x - mu;
    sm[t] = d * d; __syncthreads();
    for (int o = 128; o > 0; o >>= 1) { if (t < o) sm[t] += sm[t + o]; __syncthreads(); }
    const float var = sm[0] * (1.f / 256.f);
    Y[base + t] = d * rsqrtf(var + 1e-5f) * g[t] + b[t];
}

__global__ __launch_bounds__(256)
void k_affine(float* __restrict__ h, const float* __restrict__ g,
              const float* __restrict__ b, float scale, int H, size_t n)
{
    const size_t i = (size_t)blockIdx.x * blockDim.x + threadIdx.x;
    if (i >= n) return;
    const int c = (int)(i % H);
    h[i] = h[i] * (scale * g[c]) + b[c];
}

__global__ __launch_bounds__(256)
void k_mul(const float* __restrict__ x, const float* __restrict__ y,
           float* __restrict__ z, size_t n)
{
    const size_t i = (size_t)blockIdx.x * blockDim.x + threadIdx.x;
    if (i < n) z[i] = x[i] * y[i];
}

__global__ __launch_bounds__(256)
void k_rowdot_sigmoid(const float* __restrict__ Hm, const float* __restrict__ w,
                      const float* __restrict__ b2, float* __restrict__ out)
{
    __shared__ float sm[256];
    const int t = threadIdx.x;
    const size_t base = (size_t)blockIdx.x * 256;
    sm[t] = Hm[base + t] * w[t]; __syncthreads();
    for (int o = 128; o > 0; o >>= 1) { if (t < o) sm[t] += sm[t + o]; __syncthreads(); }
    if (t == 0) out[blockIdx.x] = 1.f / (1.f + expf(-(sm[0] + b2[0])));
}

// ---------------------------------------------------------------------------
extern "C" void kernel_launch(void* const* d_in, const int* in_sizes, int n_in,
                              void* d_out, int out_size, void* d_ws, size_t ws_size,
                              hipStream_t stream)
{
    (void)in_sizes; (void)n_in; (void)out_size; (void)ws_size;

    const float* p_feat  = (const float*)d_in[0];
    const float* t_feat  = (const float*)d_in[1];
    const float* m_meta  = (const float*)d_in[2];
    const float* me1_W   = (const float*)d_in[3];
    const float* me1_b   = (const float*)d_in[4];
    const float* me2_W   = (const float*)d_in[5];
    const float* me2_b   = (const float*)d_in[6];
    const float* sage_Wl = (const float*)d_in[7];
    const float* sage_bl = (const float*)d_in[8];
    const float* sage_Wr = (const float*)d_in[9];
    const float* lpm_W   = (const float*)d_in[10];
    const float* lpm_b   = (const float*)d_in[11];
    const float* ln_g    = (const float*)d_in[12];
    const float* ln_b    = (const float*)d_in[13];
    const float* pmA_W   = (const float*)d_in[14];
    const float* pmA_b   = (const float*)d_in[15];
    const float* pmB_W   = (const float*)d_in[16];
    const float* pmB_b   = (const float*)d_in[17];
    const float* mt1_W   = (const float*)d_in[18];
    const float* mt1_b   = (const float*)d_in[19];
    const float* bn_g    = (const float*)d_in[20];
    const float* bn_b    = (const float*)d_in[21];
    const float* mt2_W   = (const float*)d_in[22];
    const float* mt2_b   = (const float*)d_in[23];
    const float* cr1_W   = (const float*)d_in[24];
    const float* cr1_b   = (const float*)d_in[25];
    const float* cr2_W   = (const float*)d_in[26];
    const float* cr2_b   = (const float*)d_in[27];
    const int* e_pm_src  = (const int*)d_in[31];
    const int* e_pm_dst  = (const int*)d_in[32];
    const int* e_mp_src  = (const int*)d_in[33];
    const int* e_mp_dst  = (const int*)d_in[34];
    const int* e_mt_src  = (const int*)d_in[35];
    const int* e_mt_dst  = (const int*)d_in[36];
    const int* e_tm_src  = (const int*)d_in[37];
    const int* e_tm_dst  = (const int*)d_in[38];
    const int* pmt       = (const int*)d_in[39];
    const int* neg_m     = (const int*)d_in[40];
    const int* neg_t     = (const int*)d_in[41];

    // ---- workspace carving ----
    float* wsf = (float*)d_ws;
    size_t off = 0;
    auto take = [&](size_t n) { float* p = wsf + off; off += n; return p; };
    float* m_hidden = take((size_t)M_N * H_N);
    float* m_tbl    = take((size_t)M_N * D_N);
    float* aggbuf   = take((size_t)M_N * D_N);
    float* c_tm     = take((size_t)M_N * D_N);
    float* c_mp     = take((size_t)P_N * D_N);
    float* c_mt     = take((size_t)T_N * D_N);
    float* pm_res   = take((size_t)B_N * D_N);
    float* pm_h     = take((size_t)B_N * D_N);
    float* pm_tmp   = take((size_t)B_N * D_N);
    float* mt_h     = take((size_t)B_N * H_N);
    float* mt_out   = take((size_t)B_N * D_N);
    float* c_pm     = m_hidden;

    // packed f16 weights (3.3 MB; L2-resident)
    _Float16* wph = (_Float16*)take(819200);
    size_t ho = 0;
    auto takeh = [&](size_t n) { _Float16* p = wph + ho; ho += n; return p; };
    _Float16* wp_me1 = takeh(131072);
    _Float16* wp_me2 = takeh(65536);
    _Float16* wp_sWl = takeh(262144);
    _Float16* wp_sWr = takeh(262144);
    _Float16* wp_lpm = takeh(131072);
    _Float16* wp_pmA = takeh(262144);
    _Float16* wp_pmB = takeh(262144);
    _Float16* wp_mt1 = takeh(131072);
    _Float16* wp_mt2 = takeh(65536);
    _Float16* wp_cr1 = takeh(65536);

    const dim3 blk(256);
    auto pack = [&](const float* W, _Float16* Wp, int K, int N) {
        const int tot = K * N;
        k_pack_w<<<(tot + 255) / 256, blk, 0, stream>>>(W, Wp, K, N);
    };
    pack(me1_W,   wp_me1, FM_N, H_N);
    pack(me2_W,   wp_me2, H_N, D_N);
    pack(sage_Wl, wp_sWl, 4 * D_N, D_N);
    pack(sage_Wr, wp_sWr, 4 * D_N, D_N);
    pack(lpm_W,   wp_lpm, 2 * D_N, D_N);
    pack(pmA_W,   wp_pmA, 4 * D_N, D_N);
    pack(pmB_W,   wp_pmB, 4 * D_N, D_N);
    pack(mt1_W,   wp_mt1, 2 * D_N, H_N);
    pack(mt2_W,   wp_mt2, H_N, D_N);
    pack(cr1_W,   wp_cr1, D_N, H_N);

    auto gemm = [&](const float* A0, const int* i0, int i0s, int ld0,
                    const float* A1, const int* i1, int i1s, int ld1,
                    const _Float16* Wp0, const _Float16* Wp1, const float* bias,
                    const float* Res, float* C, int Mr, int N, int K0, int K1, int act) {
        dim3 grid(N / 64, (Mr + 255) / 256);
        const size_t smem = (size_t)((K0 + K1) / 32) * 4096;   // <= 64 KB
        k_wmma_gemm<<<grid, blk, smem, stream>>>(A0, i0, i0s, ld0, A1, i1, i1s, ld1,
                                                 Wp0, Wp1, bias, Res, C, Mr, N, K0, K1, act);
    };

    // 1) m_tbl = relu(m_meta @ me1_W + b) @ me2_W + b
    gemm(m_meta, nullptr, 0, FM_N, nullptr, nullptr, 0, 0,
         wp_me1, nullptr, me1_b, nullptr, m_hidden, M_N, H_N, FM_N, 0, 1);
    gemm(m_hidden, nullptr, 0, H_N, nullptr, nullptr, 0, 0,
         wp_me2, nullptr, me2_b, nullptr, m_tbl, M_N, D_N, H_N, 0, 0);

    // 2) four SAGE convs (fused K=512 dual GEMM)
    auto sage = [&](const float* srcF, const int* es, const int* ed, int n_dst,
                    const float* dstF, int w, float* cbuf) {
        const size_t n = (size_t)n_dst * D_N;
        k_fill<<<(unsigned)((n + 255) / 256), blk, 0, stream>>>(aggbuf, 0.f, n);
        k_scatter_add<<<E_N, blk, 0, stream>>>(srcF, es, ed, aggbuf);
        gemm(aggbuf, nullptr, 0, D_N, dstF, nullptr, 0, D_N,
             wp_sWl + (size_t)w * 65536, wp_sWr + (size_t)w * 65536,
             sage_bl + (size_t)w * D_N, nullptr, cbuf, n_dst, D_N, D_N, D_N, 0);
    };
    sage(p_feat, e_pm_src, e_pm_dst, M_N, m_tbl, 0, c_pm);
    sage(m_tbl,  e_mp_src, e_mp_dst, P_N, p_feat, 1, c_mp);
    sage(m_tbl,  e_mt_src, e_mt_dst, T_N, t_feat, 2, c_mt);
    sage(t_feat, e_tm_src, e_tm_dst, M_N, m_tbl, 3, c_tm);

    // 3) out_* = 0.01*tanh(.) + emb (in place)
    const size_t nPD = (size_t)P_N * D_N, nMD = (size_t)M_N * D_N, nTD = (size_t)T_N * D_N;
    k_tanh_comb<<<(unsigned)((nPD + 255) / 256), blk, 0, stream>>>(c_mp, nullptr, p_feat, c_mp, 1.f, nPD);
    k_tanh_comb<<<(unsigned)((nTD + 255) / 256), blk, 0, stream>>>(c_mt, nullptr, t_feat, c_mt, 1.f, nTD);
    k_tanh_comb<<<(unsigned)((nMD + 255) / 256), blk, 0, stream>>>(c_pm, c_tm, m_tbl, c_pm, 0.5f, nMD);
    const float* out_p = c_mp;
    const float* out_m = c_pm;
    const float* out_t = c_mt;

    // 4) scoring: set 0 = positive, 1..4 = negatives
    float* out = (float*)d_out;
    const float bn_scale = 1.0f / sqrtf(1.0f + 1e-5f);
    const size_t nBD = (size_t)B_N * D_N, nBH = (size_t)B_N * H_N;
    for (int s = 0; s < 5; ++s) {
        const int* midx; int ms; const int* tidx; int ts;
        if (s == 0) { midx = pmt + 1; ms = 3; tidx = pmt + 2; ts = 3; }
        else {
            midx = neg_m + (size_t)(s - 1) * B_N; ms = 1;
            tidx = neg_t + (size_t)(s - 1) * B_N; ts = 1;
        }

        // pm_learn
        gemm(out_p, pmt, 3, D_N, out_m, midx, ms, D_N,
             wp_lpm, wp_lpm + 65536, lpm_b, nullptr, pm_tmp, B_N, D_N, D_N, D_N, 0);
        k_layernorm<<<B_N, blk, 0, stream>>>(pm_tmp, ln_g, ln_b, pm_res);
        for (int i = 0; i < 4; ++i) {
            gemm(pm_res, nullptr, 0, D_N, nullptr, nullptr, 0, 0,
                 wp_pmA + (size_t)i * 65536, nullptr, pmA_b + (size_t)i * D_N,
                 nullptr, pm_h, B_N, D_N, D_N, 0, 2);
            gemm(pm_h, nullptr, 0, D_N, nullptr, nullptr, 0, 0,
                 wp_pmB + (size_t)i * 65536, nullptr, pmB_b + (size_t)i * D_N,
                 pm_res, pm_tmp, B_N, D_N, D_N, 0, 0);
            k_layernorm<<<B_N, blk, 0, stream>>>(pm_tmp, ln_g, ln_b, pm_res);
        }

        // mt_learn
        gemm(out_m, midx, ms, D_N, out_t, tidx, ts, D_N,
             wp_mt1, wp_mt1 + 65536, mt1_b, nullptr, mt_h, B_N, H_N, D_N, D_N, 1);
        k_affine<<<(unsigned)((nBH + 255) / 256), blk, 0, stream>>>(mt_h, bn_g, bn_b, bn_scale, H_N, nBH);
        gemm(mt_h, nullptr, 0, H_N, nullptr, nullptr, 0, 0,
             wp_mt2, nullptr, mt2_b, nullptr, mt_out, B_N, D_N, H_N, 0, 0);

        // cross
        k_mul<<<(unsigned)((nBD + 255) / 256), blk, 0, stream>>>(pm_res, mt_out, pm_tmp, nBD);
        gemm(pm_tmp, nullptr, 0, D_N, nullptr, nullptr, 0, 0,
             wp_cr1, nullptr, cr1_b, nullptr, pm_h, B_N, H_N, D_N, 0, 0);
        k_rowdot_sigmoid<<<B_N, blk, 0, stream>>>(pm_h, cr2_W, cr2_b, out + (size_t)s * B_N);
    }
}
